// S4D_78520592106103
// MI455X (gfx1250) — matrix operand order repaired and checked by reading, not compiled, not run on hardware
//
#include <hip/hip_runtime.h>
#include <math.h>

typedef __attribute__((ext_vector_type(16))) __bf16 v16bf;
typedef __attribute__((ext_vector_type(8)))  __bf16 v8bf;
typedef __attribute__((ext_vector_type(8)))  float  v8f;

#define B_     16
#define L_     4096
#define H_     512
#define NH_    32
#define OUT_   256
#define NPRED_ 128
#define ZROWS_ 256          // mix rows actually needed: 384..511 and 896..1023
#define LP_    (L_ + 128)   // krev row stride (zero-padded tail)

// ---------------------------------------------------------------- WMMA helpers
__device__ __forceinline__ v8f wmma_bf16(v16bf a, v16bf b, v8f c) {
  return __builtin_amdgcn_wmma_f32_16x16x32_bf16(false, a, false, b, (short)0, c,
                                                 false, false);
}
__device__ __forceinline__ v8bf load8a(const __bf16* p) {   // 16B-aligned b128
  return *(const v8bf*)p;
}
__device__ __forceinline__ v8bf load8u(const __bf16* p) {   // unaligned 16B
  v8bf r;
  __builtin_memcpy(&r, p, 16);
  return r;
}
__device__ __forceinline__ v16bf cat8(v8bf lo, v8bf hi) {
  return __builtin_shufflevector(lo, hi, 0, 1, 2, 3, 4, 5, 6, 7,
                                 8, 9, 10, 11, 12, 13, 14, 15);
}
__device__ __forceinline__ int zrow_map(int m) {  // 0..127->384..511 ; 128..255->896..1023
  return (m < 128) ? (384 + m) : (768 + m);
}

// ------------------- K0: (B,L,H) f32 -> (B,H,L) bf16, LDS-tiled transpose
__global__ void k_transpose(const float* __restrict__ inp, __bf16* __restrict__ xbf) {
  __shared__ float tile[32][33];
  int h0 = blockIdx.x * 32, l0 = blockIdx.y * 32, b = blockIdx.z;
  int tx = threadIdx.x, ty = threadIdx.y;  // 32 x 8
  const float* src = inp + ((size_t)b * L_ + l0) * H_ + h0;
#pragma unroll
  for (int i = 0; i < 4; ++i)
    tile[ty + 8 * i][tx] = src[(size_t)(ty + 8 * i) * H_ + tx];
  __syncthreads();
  __bf16* dst = xbf + ((size_t)b * H_ + h0) * L_ + l0;
#pragma unroll
  for (int i = 0; i < 4; ++i)
    dst[(size_t)(ty + 8 * i) * L_ + tx] = (__bf16)tile[tx][ty + 8 * i];
}

// ------------------- K0b: y (B,H,L) bf16 -> yT (B,L,H) bf16, LDS-tiled transpose
__global__ void k_transposeT(const __bf16* __restrict__ y, __bf16* __restrict__ yT) {
  __shared__ __bf16 tile[32][34];
  int h0 = blockIdx.x * 32, l0 = blockIdx.y * 32, b = blockIdx.z;
  int tx = threadIdx.x, ty = threadIdx.y;  // 32 x 8
  const __bf16* src = y + ((size_t)b * H_ + h0) * L_ + l0;
#pragma unroll
  for (int i = 0; i < 4; ++i)
    tile[ty + 8 * i][tx] = src[(size_t)(ty + 8 * i) * L_ + tx];
  __syncthreads();
  __bf16* dst = yT + ((size_t)b * L_ + l0) * H_ + h0;
#pragma unroll
  for (int i = 0; i < 4; ++i)
    dst[(size_t)(ty + 8 * i) * H_ + tx] = tile[tx][ty + 8 * i];
}

// ------------------- K1: SSM kernel, stored REVERSED with zero pad: kr[i]=k[L-1-i]
__global__ void k_ssm_kernel(const float* __restrict__ C, const float* __restrict__ log_dt,
                             const float* __restrict__ log_A_real,
                             const float* __restrict__ A_imag, __bf16* __restrict__ kr) {
  int idx = blockIdx.x * blockDim.x + threadIdx.x;  // h*LP_ + i
  int h = idx / LP_, i = idx % LP_;
  if (i >= L_) { kr[idx] = (__bf16)0.0f; return; }  // causal zero pad
  int l = L_ - 1 - i;
  float dt = expf(log_dt[h]);
  float fl = (float)l, acc = 0.f;
  for (int n = 0; n < NH_; ++n) {
    float Ar = -expf(log_A_real[h * NH_ + n]);
    float Ai = A_imag[h * NH_ + n];
    float dr = Ar * dt, di = Ai * dt;
    float ed = expf(dr);
    float er = ed * cosf(di) - 1.0f;
    float ei = ed * sinf(di);
    float inv = 1.0f / (Ar * Ar + Ai * Ai);
    float qr = (er * Ar + ei * Ai) * inv;
    float qi = (ei * Ar - er * Ai) * inv;
    float Cr = C[(h * NH_ + n) * 2 + 0], Ci = C[(h * NH_ + n) * 2 + 1];
    float wr = Cr * qr - Ci * qi;
    float wi = Cr * qi + Ci * qr;
    acc += expf(dr * fl) * (wr * cosf(di * fl) - wi * sinf(di * fl));
  }
  kr[idx] = (__bf16)(2.0f * acc);
}

// ------------------- Kc: weight conversions to bf16
__global__ void k_convert(const float* __restrict__ conv_w, const float* __restrict__ out_w,
                          __bf16* __restrict__ wbf, __bf16* __restrict__ owbf) {
  int idx = blockIdx.x * blockDim.x + threadIdx.x;
  const int NW = 2 * H_ * H_;  // 524288
  const int NO = OUT_ * L_;    // 1048576
  if (idx < NW) wbf[idx] = (__bf16)conv_w[idx];
  int j = idx - NW;
  if (j >= 0 && j < NO) owbf[j] = (__bf16)out_w[j];
}

// ------------------- K2: causal Toeplitz conv + D*x + GELU (WMMA, branch-free)
// wave = 16(batch) x 64(l) tile for one h; 4 WMMAs share one A fragment per K-step.
__global__ void __launch_bounds__(128)
k_conv(const __bf16* __restrict__ xbf, const __bf16* __restrict__ kr,
       const float* __restrict__ Dv, __bf16* __restrict__ ybf) {
  int tid = threadIdx.x, wave = tid >> 5, lane = tid & 31;
  int h = blockIdx.y;
  int lw = blockIdx.x * 256 + wave * 64;  // wave's l base (4 sub-tiles of 16)
  int col = lane & 15, hi = lane >> 4;
  const __bf16* xrow = xbf + ((size_t)col * H_ + h) * L_;  // A row: batch = lane&15
  const __bf16* krow = kr + (size_t)h * LP_;
  v8f acc[4] = {};
  int b0 = (L_ - 1) - lw - 48 - col + hi * 16;  // krev window base (>= 0 always)
  int nsteps = (lw >> 5) + 2;                   // covers u in [0, lw+63]
  const __bf16* xp = xrow + hi * 16;
  for (int s = 0; s < nsteps; ++s) {
    __builtin_prefetch(xp + 128, 0, 0);  // global_prefetch_b8
    v16bf A = cat8(load8a(xp), load8a(xp + 8));
    const __bf16* kp = krow + b0;
    v8bf c0 = load8u(kp),      c1 = load8u(kp + 8),  c2 = load8u(kp + 16),
         c3 = load8u(kp + 24), c4 = load8u(kp + 32), c5 = load8u(kp + 40),
         c6 = load8u(kp + 48), c7 = load8u(kp + 56);
    acc[0] = wmma_bf16(A, cat8(c6, c7), acc[0]);
    acc[1] = wmma_bf16(A, cat8(c4, c5), acc[1]);
    acc[2] = wmma_bf16(A, cat8(c2, c3), acc[2]);
    acc[3] = wmma_bf16(A, cat8(c0, c1), acc[3]);
    xp += 32;
    b0 += 32;
  }
  float Dh = Dv[h];
#pragma unroll
  for (int n = 0; n < 4; ++n) {
    int l = lw + 16 * n + col;
#pragma unroll
    for (int r = 0; r < 8; ++r) {
      int bb = r + hi * 8;  // M = batch
      size_t off = ((size_t)bb * H_ + h) * L_ + l;
      float yv = acc[n][r] + Dh * (float)xbf[off];
      float g = 0.5f * yv * (1.0f + erff(yv * 0.70710678118f));  // exact GELU
      ybf[off] = (__bf16)g;
    }
  }
}

// ------------------- K3: z = conv_w[needed rows] @ y + conv_b (WMMA, all b128)
__global__ void __launch_bounds__(128)
k_mix(const __bf16* __restrict__ wbf, const __bf16* __restrict__ yT,
      const float* __restrict__ conv_b, float* __restrict__ zbuf) {
  int tid = threadIdx.x, wave = tid >> 5, lane = tid & 31;
  int b = blockIdx.z;
  int m0 = blockIdx.y * 16;
  int lo = blockIdx.x * 64 + wave * 16;
  int col = lane & 15, hi = lane >> 4;
  const __bf16* arow = wbf + (size_t)zrow_map(m0 + col) * H_;       // A: channel row
  const __bf16* brow = yT + ((size_t)b * L_ + lo + col) * H_;       // B: contiguous in h
  v8f acc = {};
  for (int h0 = 0; h0 < H_; h0 += 32) {
    v16bf A = cat8(load8a(arow + h0 + hi * 16), load8a(arow + h0 + hi * 16 + 8));
    v16bf Bm = cat8(load8a(brow + h0 + hi * 16), load8a(brow + h0 + hi * 16 + 8));
    acc = wmma_bf16(A, Bm, acc);
  }
  int l = lo + col;
#pragma unroll
  for (int r = 0; r < 8; ++r) {
    int m = m0 + r + hi * 8;
    zbuf[((size_t)b * ZROWS_ + m) * L_ + l] = acc[r] + conv_b[zrow_map(m)];
  }
}

// ------------------- K4: GLU gate a * sigmoid(b) -> bf16
__global__ void k_gate(const float* __restrict__ zbuf, __bf16* __restrict__ gbf) {
  int idx = blockIdx.x * blockDim.x + threadIdx.x;  // b*128*L + j*L + l
  int l = idx & (L_ - 1);
  int t = idx >> 12;
  int j = t & (NPRED_ - 1);
  int b = t >> 7;
  float za = zbuf[((size_t)b * ZROWS_ + j) * L_ + l];
  float zb = zbuf[((size_t)b * ZROWS_ + NPRED_ + j) * L_ + l];
  gbf[idx] = (__bf16)(za * (1.0f / (1.0f + expf(-zb))));
}

// ------------------- K5: out[b,o,j] = sum_l g[b,j,l] * out_w[o,l] + out_b (WMMA, all b128)
__global__ void __launch_bounds__(128)
k_out(const __bf16* __restrict__ gbf, const __bf16* __restrict__ owbf,
      const float* __restrict__ out_b, float* __restrict__ outp) {
  int tid = threadIdx.x, wave = tid >> 5, lane = tid & 31;
  int b = blockIdx.z;
  int j0 = blockIdx.y * 16;
  int o0 = (blockIdx.x * 4 + wave) * 16;
  int col = lane & 15, hi = lane >> 4;
  const __bf16* arow = gbf + ((size_t)b * NPRED_ + j0 + col) * L_;  // A: h-slice row
  const __bf16* brow = owbf + (size_t)(o0 + col) * L_;              // B: output col
  v8f acc = {};
  for (int l0 = 0; l0 < L_; l0 += 32) {
    v16bf A = cat8(load8a(arow + l0 + hi * 16), load8a(arow + l0 + hi * 16 + 8));
    v16bf Bm = cat8(load8a(brow + l0 + hi * 16), load8a(brow + l0 + hi * 16 + 8));
    acc = wmma_bf16(A, Bm, acc);
  }
  int o = o0 + col;
  float bias = out_b[o];
#pragma unroll
  for (int r = 0; r < 8; ++r) {
    int j = j0 + r + hi * 8;
    outp[((size_t)b * OUT_ + o) * NPRED_ + j] = acc[r] + bias;
  }
}

// ------------------------------------------------------------------ launcher
extern "C" void kernel_launch(void* const* d_in, const int* in_sizes, int n_in,
                              void* d_out, int out_size, void* d_ws, size_t ws_size,
                              hipStream_t stream) {
  const float* inp        = (const float*)d_in[0];
  // d_in[1] = spec : unused by the reference
  const float* Cm         = (const float*)d_in[2];
  const float* log_dt     = (const float*)d_in[3];
  const float* log_A_real = (const float*)d_in[4];
  const float* A_imag     = (const float*)d_in[5];
  const float* Dv         = (const float*)d_in[6];
  const float* conv_w     = (const float*)d_in[7];
  const float* conv_b     = (const float*)d_in[8];
  const float* out_w      = (const float*)d_in[9];
  const float* out_b      = (const float*)d_in[10];
  float* outp = (float*)d_out;

  char* ws = (char*)d_ws;
  size_t off = 0;
  auto alloc = [&](size_t bytes) -> void* {
    void* p = ws + off;
    off += (bytes + 255) & ~(size_t)255;
    return p;
  };
  __bf16* xbf  = (__bf16*)alloc((size_t)B_ * H_ * L_ * 2);      // 64 MB
  __bf16* ybf  = (__bf16*)alloc((size_t)B_ * H_ * L_ * 2);      // 64 MB
  __bf16* yT   = (__bf16*)alloc((size_t)B_ * L_ * H_ * 2);      // 64 MB
  __bf16* kr   = (__bf16*)alloc((size_t)H_ * LP_ * 2);          // 4.1 MB (reversed+pad)
  __bf16* wbf  = (__bf16*)alloc((size_t)2 * H_ * H_ * 2);       // 1 MB
  __bf16* owbf = (__bf16*)alloc((size_t)OUT_ * L_ * 2);         // 2 MB
  float*  zbuf = (float*) alloc((size_t)B_ * ZROWS_ * L_ * 4);  // 64 MB
  __bf16* gbf  = (__bf16*)alloc((size_t)B_ * NPRED_ * L_ * 2);  // 16 MB

  k_transpose<<<dim3(H_ / 32, L_ / 32, B_), dim3(32, 8), 0, stream>>>(inp, xbf);
  k_ssm_kernel<<<(H_ * LP_) / 256, 256, 0, stream>>>(Cm, log_dt, log_A_real, A_imag, kr);
  k_convert<<<(2 * H_ * H_ + OUT_ * L_) / 256, 256, 0, stream>>>(conv_w, out_w, wbf, owbf);
  k_conv<<<dim3(L_ / 256, H_), 128, 0, stream>>>(xbf, kr, Dv, ybf);
  k_transposeT<<<dim3(H_ / 32, L_ / 32, B_), dim3(32, 8), 0, stream>>>(ybf, yT);
  k_mix<<<dim3(L_ / 64, ZROWS_ / 16, B_), 128, 0, stream>>>(wbf, yT, conv_b, zbuf);
  k_gate<<<(B_ * NPRED_ * L_) / 256, 256, 0, stream>>>(zbuf, gbf);
  k_out<<<dim3(OUT_ / 64, NPRED_ / 16, B_), 128, 0, stream>>>(gbf, owbf, out_b, outp);
}
